// MultiHeadAttention_29678224015584
// MI455X (gfx1250) — matrix operand order repaired and checked by reading, not compiled, not run on hardware
//
#include <hip/hip_runtime.h>

// ---------------------------------------------------------------------------
// MHA forward for gfx1250 (MI455X), wave32.
//  * every matmul through v_wmma_f32_16x16x32_f16 (f16 in, f32 accum)
//  * GEMMs: 32x64 per-wave register tiles (8 WMMAs per 12 b128 loads)
//  * flash attention: K/V tiles staged in LDS by the Tensor Data Mover
//    (tensor_load_to_lds + s_wait_tensorcnt), double-buffered, shared by
//    all 8 waves of a block -> 8x less L2 traffic than per-wave streaming.
// ---------------------------------------------------------------------------

typedef __attribute__((ext_vector_type(16))) _Float16 v16h;
typedef __attribute__((ext_vector_type(8)))  float    v8f;
typedef __attribute__((ext_vector_type(4)))  unsigned int v4u;
typedef __attribute__((ext_vector_type(4)))  unsigned int u32x4;
typedef __attribute__((ext_vector_type(8)))  int      i32x8;
typedef __attribute__((ext_vector_type(4)))  int      i32x4;

#define D_MODEL   1024
#define SEQ       2048
#define BATCH     2
#define NHEAD     16
#define HDIM      64
#define ROWS      (BATCH * SEQ)          // 4096 flat rows
#define KTILE     32                     // keys per flash iteration

// ----- fragment helpers -----------------------------------------------------
// A fragment (16x32 f16, M x K): lanes 0-15 hold M=0..15 with K {0..7,16..23};
// lanes 16-31 same M with K {8..15,24..31}.
__device__ __forceinline__ v16h load_a_f16(const _Float16* __restrict__ base,
                                           int ld, int row0, int k0, int lane) {
    int r = row0 + (lane & 15);
    int g = lane >> 4;
    const _Float16* p = base + (size_t)r * ld + k0 + g * 8;
    union { v16h h; v4u u[2]; } f;
    f.u[0] = *(const v4u*)p;          // K = k0+8g    .. +7
    f.u[1] = *(const v4u*)(p + 16);   // K = k0+16+8g .. +7
    return f.h;
}

// B fragment (32x16 f16, K x N) where B[k][n] = rows[n][k] of a row-major
// matrix. Lanes 0-15 hold K {0..7,16..23}, lanes 16-31 K {8..15,24..31}.
__device__ __forceinline__ v16h load_b_rows_f16(const _Float16* __restrict__ base,
                                                int ld, int n0, int k0, int lane) {
    int n = n0 + (lane & 15);
    int g = lane >> 4;
    const _Float16* p = base + (size_t)n * ld + k0 + g * 8;
    union { v16h h; v4u u[2]; } f;
    f.u[0] = *(const v4u*)p;
    f.u[1] = *(const v4u*)(p + 16);
    return f.h;
}

__device__ __forceinline__ v8f wmma_f16(v16h a, v16h b, v8f c) {
    return __builtin_amdgcn_wmma_f32_16x16x32_f16(
        false, a, false, b, (short)0, c, false, false);
}

// ----- Tensor Data Mover: 2-D tile DMA global -> LDS (ISA 08 s8) -----------
// Dims/strides in elements (data_size = 2 bytes). Groups 2/3 zero (2-D).
__device__ __forceinline__ void tdm_load_2d(unsigned lds_byte_off,
                                            const void* gaddr,
                                            unsigned tensor_d0, unsigned tensor_d1,
                                            unsigned tile_d0,   unsigned tile_d1,
                                            unsigned stride0) {
    unsigned long long ga = (unsigned long long)(size_t)gaddr;
    u32x4 g0;
    g0[0] = 1u;                                            // count=1 (user D#)
    g0[1] = lds_byte_off;                                  // lds_addr
    g0[2] = (unsigned)(ga & 0xFFFFFFFFu);                  // global_addr[31:0]
    g0[3] = (unsigned)((ga >> 32) & 0x1FFFFFFu) | (2u << 30); // [56:32] | type=2
    i32x8 g1;
    g1[0] = (int)(1u << 16);                               // data_size=1 (2B)
    g1[1] = (int)((tensor_d0 & 0xFFFFu) << 16);            // tensor_dim0 lo
    g1[2] = (int)((tensor_d0 >> 16) | ((tensor_d1 & 0xFFFFu) << 16));
    g1[3] = (int)((tensor_d1 >> 16) | (tile_d0 << 16));    // tile_dim0
    g1[4] = (int)tile_d1;                                  // tile_dim1 (dim2=0)
    g1[5] = (int)stride0;                                  // dim0_stride[31:0]
    g1[6] = 0;
    g1[7] = 0;
    i32x4 z4 = {0, 0, 0, 0};
    i32x8 z8 = {0, 0, 0, 0, 0, 0, 0, 0};
    // 6-arg variant (clang-23 / therock-10.0 headers)
    __builtin_amdgcn_tensor_load_to_lds(g0, g1, z4, z4, z8, 0);
}

__device__ __forceinline__ unsigned lds_off(const void* p) {
    // generic LDS address: low 32 bits are the LDS byte offset (ISA 10.2)
    return (unsigned)(unsigned long long)(size_t)p;
}

// ----- fp32 -> fp16 convert -------------------------------------------------
__global__ void cvt_f32_to_f16(const float* __restrict__ src,
                               _Float16* __restrict__ dst, int n) {
    int i = blockIdx.x * blockDim.x + threadIdx.x;
    if (i < n) dst[i] = (_Float16)src[i];
}

// ----- QKV projection: 32x64 tile/wave -> Q/K [b,h,s,hd], V^T [b,h,hd,s] ----
__global__ void qkv_proj_kernel(const _Float16* __restrict__ xh,
                                const _Float16* __restrict__ Wqh,
                                const _Float16* __restrict__ Wkh,
                                const _Float16* __restrict__ Wvh,
                                _Float16* __restrict__ Qb,
                                _Float16* __restrict__ Kb,
                                _Float16* __restrict__ Vt) {
    int lane = threadIdx.x & 31;
    int wave = threadIdx.x >> 5;
    int rowTile = blockIdx.x * 8 + wave;   // 0..127 (32 rows each)
    int h    = blockIdx.y;
    int proj = blockIdx.z;                 // 0=Q 1=K 2=V
    const _Float16* W = (proj == 0) ? Wqh : (proj == 1) ? Wkh : Wvh;
    int r0 = rowTile * 32;

    v8f acc[2][4] = {};                    // 2 row-groups x 4 col-frags
    for (int kk = 0; kk < D_MODEL; kk += 32) {
        v16h a0 = load_a_f16(xh, D_MODEL, r0,      kk, lane);
        v16h a1 = load_a_f16(xh, D_MODEL, r0 + 16, kk, lane);
#pragma unroll
        for (int c = 0; c < 4; ++c) {
            v16h bf = load_b_rows_f16(W, D_MODEL, h * HDIM + c * 16, kk, lane);
            acc[0][c] = wmma_f16(a0, bf, acc[0][c]);
            acc[1][c] = wmma_f16(a1, bf, acc[1][c]);
        }
    }

    int g = lane >> 4, n = lane & 15;
#pragma unroll
    for (int rg = 0; rg < 2; ++rg)
#pragma unroll
        for (int c = 0; c < 4; ++c)
#pragma unroll
            for (int i = 0; i < 8; ++i) {
                int row = r0 + rg * 16 + i + 8 * g;    // flat b*2048+s
                int b   = row >> 11, s = row & 2047;
                int hd  = c * 16 + n;
                _Float16 v = (_Float16)acc[rg][c][i];
                if (proj == 2)
                    Vt[(((size_t)b * NHEAD + h) * HDIM + hd) * SEQ + s] = v;
                else if (proj == 0)
                    Qb[(((size_t)b * NHEAD + h) * SEQ + s) * HDIM + hd] = v;
                else
                    Kb[(((size_t)b * NHEAD + h) * SEQ + s) * HDIM + hd] = v;
            }
}

// ----- flash attention: TDM-staged K/V in LDS, one wave = 16 queries --------
__global__ void flash_attn_kernel(const _Float16* __restrict__ Qb,
                                  const _Float16* __restrict__ Kb,
                                  const _Float16* __restrict__ Vt,
                                  _Float16* __restrict__ ctxh) {
    __shared__ __attribute__((aligned(128))) _Float16 shK[2][KTILE * HDIM]; // 32x64
    __shared__ __attribute__((aligned(128))) _Float16 shV[2][HDIM * KTILE]; // 64x32

    int lane = threadIdx.x & 31;
    int wave = threadIdx.x >> 5;
    int qt = blockIdx.x * 8 + wave;        // 8 q-tiles/block, same (b,h) plane
    int h  = blockIdx.y;
    int b  = blockIdx.z;
    const _Float16* Qp = Qb + ((size_t)(b * NHEAD + h)) * SEQ * HDIM;
    const _Float16* Kp = Kb + ((size_t)(b * NHEAD + h)) * SEQ * HDIM;
    const _Float16* Vp = Vt + ((size_t)(b * NHEAD + h)) * HDIM * SEQ;
    int q0 = qt * 16;
    int g  = lane >> 4;

    // prologue DMA: tile 0 into buffer 0 (one wave drives the TDM)
    if (wave == 0) {
        tdm_load_2d(lds_off(&shK[0][0]), Kp,        HDIM, SEQ,  HDIM,  KTILE, HDIM);
        tdm_load_2d(lds_off(&shV[0][0]), Vp,        SEQ,  HDIM, KTILE, HDIM,  SEQ);
    }

    // Q as B fragments (columns of Q^T), resident for the whole loop.
    v16h bq0 = load_b_rows_f16(Qp, HDIM, q0, 0,  lane);
    v16h bq1 = load_b_rows_f16(Qp, HDIM, q0, 32, lane);

    v8f acc[4] = {};                       // ctx: 16 queries x 64 hd
    float m = -1e30f, l = 0.f;
    const float sc = 0.03125f;             // 1/sqrt(1024)

    const int NIT = SEQ / KTILE;           // 64
    for (int it = 0; it < NIT; ++it) {
        int buf = it & 1;
        if (wave == 0) __builtin_amdgcn_s_wait_tensorcnt(0);
        __syncthreads();                   // buf ready; buf^1 free to refill
        if (wave == 0 && it + 1 < NIT) {
            int kn = (it + 1) * KTILE;
            tdm_load_2d(lds_off(&shK[buf ^ 1][0]), Kp + (size_t)kn * HDIM,
                        HDIM, SEQ,  HDIM,  KTILE, HDIM);
            tdm_load_2d(lds_off(&shV[buf ^ 1][0]), Vp + kn,
                        SEQ,  HDIM, KTILE, HDIM,  SEQ);
        }
        const _Float16* Kt = &shK[buf][0];   // 32 keys x 64 hd   (ld 64)
        const _Float16* Vb = &shV[buf][0];   // 64 hd   x 32 keys (ld 32)

        // S^T tiles: rows = keys, cols = queries
        v8f st0 = {}, st1 = {};
        v16h ak;
        ak = load_a_f16(Kt, HDIM, 0,  0,  lane);  st0 = wmma_f16(ak, bq0, st0);
        ak = load_a_f16(Kt, HDIM, 0,  32, lane);  st0 = wmma_f16(ak, bq1, st0);
        ak = load_a_f16(Kt, HDIM, 16, 0,  lane);  st1 = wmma_f16(ak, bq0, st1);
        ak = load_a_f16(Kt, HDIM, 16, 32, lane);  st1 = wmma_f16(ak, bq1, st1);

        float v0[8], v1[8], tmax = -1e30f;
#pragma unroll
        for (int i = 0; i < 8; ++i) {
            v0[i] = st0[i] * sc; v1[i] = st1[i] * sc;
            tmax = fmaxf(tmax, fmaxf(v0[i], v1[i]));
        }
        tmax = fmaxf(tmax, __shfl_xor(tmax, 16, 32));
        float mn = fmaxf(m, tmax);
        float cf = __expf(m - mn);

        float p0[8], p1[8], rs = 0.f;
#pragma unroll
        for (int i = 0; i < 8; ++i) {
            p0[i] = __expf(v0[i] - mn);
            p1[i] = __expf(v1[i] - mn);
            rs += p0[i] + p1[i];
        }
        rs += __shfl_xor(rs, 16, 32);
        l = l * cf + rs;
        m = mn;

        float cfi[8];
#pragma unroll
        for (int i = 0; i < 8; ++i) cfi[i] = __shfl(cf, i + 8 * g, 32);
#pragma unroll
        for (int c = 0; c < 4; ++c)
#pragma unroll
            for (int i = 0; i < 8; ++i) acc[c][i] *= cfi[i];

        // P^T (C layout) packs directly into the A fragment of P.
        union { v16h hh; _Float16 e[16]; } pa;
#pragma unroll
        for (int i = 0; i < 8; ++i) {
            pa.e[i]     = (_Float16)p0[i];
            pa.e[8 + i] = (_Float16)p1[i];
        }
#pragma unroll
        for (int c = 0; c < 4; ++c) {
            v16h bv = load_b_rows_f16(Vb, KTILE, c * 16, 0, lane); // V^T rows
            acc[c] = wmma_f16(pa.hh, bv, acc[c]);
        }
    }

    float linv = 1.f / l, li[8];
#pragma unroll
    for (int i = 0; i < 8; ++i) li[i] = __shfl(linv, i + 8 * g, 32);

    int n = lane & 15;
#pragma unroll
    for (int c = 0; c < 4; ++c)
#pragma unroll
        for (int i = 0; i < 8; ++i) {
            int s   = q0 + i + 8 * g;
            int col = h * HDIM + c * 16 + n;
            ctxh[((size_t)(b * SEQ + s)) * D_MODEL + col] =
                (_Float16)(acc[c][i] * li[i]);
        }
}

// ----- output projection: 32x64 tile/wave, ctx @ Wo^T -> fp32 d_out ---------
__global__ void out_proj_kernel(const _Float16* __restrict__ ctxh,
                                const _Float16* __restrict__ Woh,
                                float* __restrict__ out) {
    int lane = threadIdx.x & 31;
    int wave = threadIdx.x >> 5;
    int rowTile = blockIdx.x * 8 + wave;   // 0..127
    int cg = blockIdx.y;                   // 64-col group, 0..15
    int r0 = rowTile * 32;

    v8f acc[2][4] = {};
    for (int kk = 0; kk < D_MODEL; kk += 32) {
        v16h a0 = load_a_f16(ctxh, D_MODEL, r0,      kk, lane);
        v16h a1 = load_a_f16(ctxh, D_MODEL, r0 + 16, kk, lane);
#pragma unroll
        for (int c = 0; c < 4; ++c) {
            v16h bf = load_b_rows_f16(Woh, D_MODEL, cg * HDIM + c * 16, kk, lane);
            acc[0][c] = wmma_f16(a0, bf, acc[0][c]);
            acc[1][c] = wmma_f16(a1, bf, acc[1][c]);
        }
    }
    int g = lane >> 4, n = lane & 15;
#pragma unroll
    for (int rg = 0; rg < 2; ++rg)
#pragma unroll
        for (int c = 0; c < 4; ++c)
#pragma unroll
            for (int i = 0; i < 8; ++i) {
                int row = r0 + rg * 16 + i + 8 * g;
                out[(size_t)row * D_MODEL + cg * HDIM + c * 16 + n] =
                    acc[rg][c][i];
            }
}

// ----- launcher -------------------------------------------------------------
extern "C" void kernel_launch(void* const* d_in, const int* in_sizes, int n_in,
                              void* d_out, int out_size, void* d_ws, size_t ws_size,
                              hipStream_t stream) {
    (void)in_sizes; (void)n_in; (void)out_size; (void)ws_size;
    const float* x  = (const float*)d_in[0];
    const float* Wq = (const float*)d_in[1];
    const float* Wk = (const float*)d_in[2];
    const float* Wv = (const float*)d_in[3];
    const float* Wo = (const float*)d_in[4];

    char* ws = (char*)d_ws;
    const size_t MB = 1u << 20;
    _Float16* xh   = (_Float16*)(ws + 0 * MB);    // 8 MiB  [4096,1024]
    _Float16* Wqh  = (_Float16*)(ws + 8 * MB);    // 2 MiB
    _Float16* Wkh  = (_Float16*)(ws + 10 * MB);   // 2 MiB
    _Float16* Wvh  = (_Float16*)(ws + 12 * MB);   // 2 MiB
    _Float16* Woh  = (_Float16*)(ws + 14 * MB);   // 2 MiB
    _Float16* Qb   = (_Float16*)(ws + 16 * MB);   // 8 MiB  [b,h,s,hd]
    _Float16* Kb   = (_Float16*)(ws + 24 * MB);   // 8 MiB  [b,h,s,hd]
    _Float16* Vt   = (_Float16*)(ws + 32 * MB);   // 8 MiB  [b,h,hd,s]
    _Float16* ctxh = (_Float16*)(ws + 40 * MB);   // 8 MiB  [4096,1024]

    const int NX = ROWS * D_MODEL;
    const int NW = D_MODEL * D_MODEL;
    cvt_f32_to_f16<<<NX / 256, 256, 0, stream>>>(x,  xh,  NX);
    cvt_f32_to_f16<<<NW / 256, 256, 0, stream>>>(Wq, Wqh, NW);
    cvt_f32_to_f16<<<NW / 256, 256, 0, stream>>>(Wk, Wkh, NW);
    cvt_f32_to_f16<<<NW / 256, 256, 0, stream>>>(Wv, Wvh, NW);
    cvt_f32_to_f16<<<NW / 256, 256, 0, stream>>>(Wo, Woh, NW);

    qkv_proj_kernel<<<dim3(16, NHEAD, 3), 256, 0, stream>>>(
        xh, Wqh, Wkh, Wvh, Qb, Kb, Vt);

    flash_attn_kernel<<<dim3(16, NHEAD, BATCH), 256, 0, stream>>>(
        Qb, Kb, Vt, ctxh);

    out_proj_kernel<<<dim3(16, NHEAD, 1), 256, 0, stream>>>(
        ctxh, Woh, (float*)d_out);
}
